// SymAttention_56796647522630
// MI455X (gfx1250) — compile-verified, matrix operand order checked by reading, not run on hardware
//
#include <hip/hip_runtime.h>
#include <hip/hip_bf16.h>
#include <stdint.h>

// ---------------------------------------------------------------------------
// Types for WMMA (gfx1250, wave32)
// ---------------------------------------------------------------------------
typedef __attribute__((ext_vector_type(16))) __bf16       v16bf;
typedef __attribute__((ext_vector_type(8)))  float        v8f;
typedef __attribute__((ext_vector_type(4)))  unsigned int u32x4;
typedef __attribute__((ext_vector_type(4)))  float        f32x4;

union Frag16 { v16bf v; u32x4 q[2]; };

__device__ __forceinline__ unsigned short f32_to_bf16(float f) {
  union { float f; unsigned u; } x; x.f = f;
  unsigned u = x.u;
  u += 0x7FFFu + ((u >> 16) & 1u);   // round-to-nearest-even
  return (unsigned short)(u >> 16);
}
__device__ __forceinline__ unsigned int pack2_bf16(float lo, float hi) {
  return (unsigned int)f32_to_bf16(lo) | ((unsigned int)f32_to_bf16(hi) << 16);
}

// ---------------------------------------------------------------------------
// Weight prep kernels (f32 -> bf16)
// ---------------------------------------------------------------------------
__global__ void cvt_bf16_kernel(const float* __restrict__ src,
                                unsigned short* __restrict__ dst, int n) {
  int i = blockIdx.x * blockDim.x + threadIdx.x;
  if (i < n) dst[i] = f32_to_bf16(src[i]);
}

// Fuse offset_w[64,Cin,3,3] and attn_w[32,Cin,3,3] into [96, 9*256] bf16,
// K reordered r-major: dst[co][r*256 + ci] = w[co][ci*9 + r].  This makes
// every 32-wide K-chunk share one (dy,dx) tap -> cheap im2col gather.
__global__ void prep_offattn_w_kernel(const float* __restrict__ ow,
                                      const float* __restrict__ aw,
                                      unsigned short* __restrict__ dst) {
  int i = blockIdx.x * blockDim.x + threadIdx.x;
  const int KD = 2304;
  if (i >= 96 * KD) return;
  int row = i / KD, kk = i - row * KD;
  int r  = kk >> 8;          // 0..8 (3x3 tap)
  int ci = kk & 255;         // input channel
  int src_k = ci * 9 + r;
  float v = (row < 64) ? ow[row * KD + src_k] : aw[(row - 64) * KD + src_k];
  dst[i] = f32_to_bf16(v);
}

__global__ void prep_offattn_b_kernel(const float* __restrict__ ob,
                                      const float* __restrict__ ab,
                                      float* __restrict__ dst) {
  int i = threadIdx.x;
  if (i < 64) dst[i] = ob[i];
  else if (i < 96) dst[i] = ab[i - 64];
}

// ---------------------------------------------------------------------------
// Unified WMMA GEMM / implicit-GEMM conv.
//   out[n,co,p] = sum_k W[co,k] * B(k,p)   (+bias, +res, relu)
// IS_CONV=0: B(k,p) = X[n,k,p]            (1x1 conv, Kdim == Cin)
// IS_CONV=1: B(k,p) = X[n,ci,(y+dy)*64+(x+dx)] with k = r*256+ci, zero-pad,
//            H=W=64; every 32-chunk of k has a single (dy,dx).
// Tile: 128(M) x 128(N) x 32(K); 256 thr = 8 waves (4x2); wave = 32x64 of C
// (2x4 WMMA tiles -> 8 v_wmma per K-step).  Double-buffered LDS.
// ---------------------------------------------------------------------------
template<int IS_CONV, int RELU, int HAS_RES>
__global__ __launch_bounds__(256)
void gemm_wmma(const unsigned short* __restrict__ Wbf,  // [Cout, Kdim] bf16
               const float* __restrict__ X,             // [B, Cin, 4096]
               const float* __restrict__ bias,          // [Cout]
               const float* __restrict__ res,           // [B, Cout, 4096]|null
               float* __restrict__ out,                 // [B, Cout, 4096]
               int Cout, int Kdim, int Cin)
{
  const int P = 4096;
  __shared__ unsigned short lds_a[2][128][40];   // [buf][m][k]
  __shared__ unsigned short lds_b[2][128][40];   // [buf][p][k] (transposed)

  const int tid  = threadIdx.x;
  const int lane = tid & 31;
  const int w    = tid >> 5;
  const int wm   = w & 3, wn = w >> 2;      // wave grid 4 x 2
  const int p0   = blockIdx.x * 128;
  const int m0   = blockIdx.y * 128;
  const int n    = blockIdx.z;

  const float* Xn = X + (size_t)n * Cin * P;

  // ---- loader thread mapping ----
  const int arow = tid >> 1;                // 0..127
  const int akc  = (tid & 1) * 16;          // 0 | 16
  const int kl2  = (tid >> 4) * 2;          // 0,2,..,30 (k pair)
  const int bpc  = (tid & 15) * 8;          // 0..120    (8 pixels)

  u32x4 areg[2];
  unsigned int breg[8];

  auto load_tiles = [&](int k0) {
    // A: 2 x 16B of bf16 weights
    u32x4 z = {0u, 0u, 0u, 0u};
    areg[0] = z; areg[1] = z;
    int co = m0 + arow;
    if (co < Cout) {
      const unsigned short* ap = Wbf + (size_t)co * Kdim + k0 + akc;
      areg[0] = *(const u32x4*)(ap);
      areg[1] = *(const u32x4*)(ap + 8);
    }
    // B: 8 pixels x 2 k's, packed as bf16 pairs
    if (!IS_CONV) {
      const float* s0 = Xn + (size_t)(k0 + kl2) * P + p0 + bpc;
      const float* s1 = s0 + P;
      f32x4 a0 = *(const f32x4*)(s0);
      f32x4 a1 = *(const f32x4*)(s0 + 4);
      f32x4 b0 = *(const f32x4*)(s1);
      f32x4 b1 = *(const f32x4*)(s1 + 4);
      #pragma unroll
      for (int i = 0; i < 4; ++i) breg[i]     = pack2_bf16(a0[i], b0[i]);
      #pragma unroll
      for (int i = 0; i < 4; ++i) breg[i + 4] = pack2_bf16(a1[i], b1[i]);
    } else {
      int r  = k0 >> 8;                     // tap index, uniform per chunk
      int rd = r / 3;
      int dy = rd - 1;
      int dx = (r - rd * 3) - 1;
      int ci = (k0 & 255) + kl2;
      const float* s0 = Xn + (size_t)ci * P;
      const float* s1 = s0 + P;
      int pbase = p0 + bpc;                 // 8-chunk stays in one image row
      int y  = (pbase >> 6) + dy;
      int xb = (pbase & 63) + dx;
      bool vy = (unsigned)y < 64u;
      int rowoff = y << 6;
      #pragma unroll
      for (int i = 0; i < 8; ++i) {
        int x = xb + i;
        bool v = vy && ((unsigned)x < 64u);
        float f0 = v ? s0[rowoff + x] : 0.f;
        float f1 = v ? s1[rowoff + x] : 0.f;
        breg[i] = pack2_bf16(f0, f1);
      }
    }
  };

  auto store_tiles = [&](int buf) {
    *(u32x4*)&lds_a[buf][arow][akc]     = areg[0];
    *(u32x4*)&lds_a[buf][arow][akc + 8] = areg[1];
    #pragma unroll
    for (int i = 0; i < 8; ++i)
      *(unsigned int*)&lds_b[buf][bpc + i][kl2] = breg[i];
  };

  v8f acc[2][4];
  #pragma unroll
  for (int i = 0; i < 2; ++i)
    #pragma unroll
    for (int j = 0; j < 4; ++j) { v8f zz = {}; acc[i][j] = zz; }

  const int NT = Kdim / 32;

  // prologue
  load_tiles(0);
  store_tiles(0);
  __syncthreads();

  const int kb   = (lane >= 16) ? 8 : 0;    // half-wave k offset (ISA layout)
  const int lrow = lane & 15;

  for (int kt = 0; kt < NT; ++kt) {
    const int cur = kt & 1;
    const bool has_next = (kt + 1) < NT;
    if (has_next) load_tiles((kt + 1) * 32);   // global loads overlap WMMAs

    Frag16 a0, a1;
    const int ar0 = wm * 32 + lrow;
    a0.q[0] = *(const u32x4*)&lds_a[cur][ar0][kb];
    a0.q[1] = *(const u32x4*)&lds_a[cur][ar0][kb + 16];
    a1.q[0] = *(const u32x4*)&lds_a[cur][ar0 + 16][kb];
    a1.q[1] = *(const u32x4*)&lds_a[cur][ar0 + 16][kb + 16];

    #pragma unroll
    for (int nb = 0; nb < 4; ++nb) {
      Frag16 b;
      const int bp = wn * 64 + nb * 16 + lrow;
      b.q[0] = *(const u32x4*)&lds_b[cur][bp][kb];
      b.q[1] = *(const u32x4*)&lds_b[cur][bp][kb + 16];
      acc[0][nb] = __builtin_amdgcn_wmma_f32_16x16x32_bf16(
          false, a0.v, false, b.v, (short)0, acc[0][nb], false, false);
      acc[1][nb] = __builtin_amdgcn_wmma_f32_16x16x32_bf16(
          false, a1.v, false, b.v, (short)0, acc[1][nb], false, false);
    }

    if (has_next) store_tiles(1 - cur);
    __syncthreads();
  }

  // ---- epilogue: C layout -> lane holds col lane&15, rows r+(lane>=16?8:0)
  const int rofs = (lane >= 16) ? 8 : 0;
  const size_t nbase = (size_t)n * Cout * P;
  #pragma unroll
  for (int ma = 0; ma < 2; ++ma) {
    #pragma unroll
    for (int r = 0; r < 8; ++r) {
      int co = m0 + wm * 32 + ma * 16 + rofs + r;
      if (co < Cout) {
        float bv = bias[co];
        size_t rowb = nbase + (size_t)co * P;
        #pragma unroll
        for (int nb = 0; nb < 4; ++nb) {
          int pc = p0 + wn * 64 + nb * 16 + lrow;
          float v = acc[ma][nb][r] + bv;
          if (HAS_RES) v += res[rowb + pc];
          if (RELU)    v = fmaxf(v, 0.f);
          out[rowb + pc] = v;
        }
      }
    }
  }
}

// ---------------------------------------------------------------------------
// Softmax over K + bilinear grid-sample (zero pad) + weighted sum over K.
// One thread per (n, m, pixel); loops 32 channels of its head.
// offattn: [n][96][4096]  ch 0..63 = offsets (m*8 + k*2 + {x,y}),
//                          ch 64..95 = attn logits (64 + m*4 + k)
// ---------------------------------------------------------------------------
__global__ __launch_bounds__(256)
void sample_attn_kernel(const float* __restrict__ offattn,
                        const float* __restrict__ value,   // [n][256][4096]
                        float* __restrict__ out)           // [n][256][4096]
{
  int gid = blockIdx.x * blockDim.x + threadIdx.x;
  if (gid >= 8 * 8 * 4096) return;
  int p  = gid & 4095;
  int nm = gid >> 12;
  int m  = nm & 7;
  int n  = nm >> 3;

  const float* oa = offattn + (size_t)n * 96 * 4096;

  float l[4];
  #pragma unroll
  for (int k = 0; k < 4; ++k) l[k] = oa[(size_t)(64 + m * 4 + k) * 4096 + p];
  float mx = fmaxf(fmaxf(l[0], l[1]), fmaxf(l[2], l[3]));
  float e[4], s = 0.f;
  #pragma unroll
  for (int k = 0; k < 4; ++k) { e[k] = __expf(l[k] - mx); s += e[k]; }
  float inv = 1.f / s;

  int x = p & 63, y = p >> 6;
  float gx_sym = (float)(63 - x);

  int   idx[4][4];
  float wt[4][4];
  #pragma unroll
  for (int k = 0; k < 4; ++k) {
    float ox = oa[(size_t)((m * 4 + k) * 2 + 0) * 4096 + p];
    float oy = oa[(size_t)((m * 4 + k) * 2 + 1) * 4096 + p];
    float px = gx_sym + ox;
    float py = (float)y + oy;
    float x0f = floorf(px), y0f = floorf(py);
    float wx = px - x0f,    wy = py - y0f;
    int x0 = (int)x0f, y0 = (int)y0f;
    float ak = e[k] * inv;
    int   cx[4] = { x0, x0 + 1, x0,     x0 + 1 };
    int   cy[4] = { y0, y0,     y0 + 1, y0 + 1 };
    float cw[4] = { (1.f - wx) * (1.f - wy), wx * (1.f - wy),
                    (1.f - wx) * wy,         wx * wy };
    #pragma unroll
    for (int c = 0; c < 4; ++c) {
      bool valid = ((unsigned)cx[c] < 64u) && ((unsigned)cy[c] < 64u);
      int xc = min(max(cx[c], 0), 63);
      int yc = min(max(cy[c], 0), 63);
      idx[k][c] = (yc << 6) + xc;
      wt[k][c]  = valid ? ak * cw[c] : 0.f;
    }
  }

  const float* vb = value + (size_t)n * 256 * 4096 + (size_t)m * 32 * 4096;
  float*       ob = out   + (size_t)n * 256 * 4096 + (size_t)m * 32 * 4096;
  for (int cm = 0; cm < 32; ++cm) {
    const float* vc = vb + (size_t)cm * 4096;
    float acc = 0.f;
    #pragma unroll
    for (int k = 0; k < 4; ++k)
      #pragma unroll
      for (int c = 0; c < 4; ++c)
        acc += wt[k][c] * vc[idx[k][c]];
    ob[(size_t)cm * 4096 + p] = acc;
  }
}

// ---------------------------------------------------------------------------
// Host launch
// ---------------------------------------------------------------------------
extern "C" void kernel_launch(void* const* d_in, const int* in_sizes, int n_in,
                              void* d_out, int out_size, void* d_ws, size_t ws_size,
                              hipStream_t stream) {
  (void)in_sizes; (void)n_in; (void)out_size; (void)ws_size;

  const float* f_recalib = (const float*)d_in[0];
  const float* f_orig    = (const float*)d_in[1];
  const float* offset_w  = (const float*)d_in[2];
  const float* offset_b  = (const float*)d_in[3];
  const float* attn_w    = (const float*)d_in[4];
  const float* attn_b    = (const float*)d_in[5];
  const float* value_w   = (const float*)d_in[6];
  const float* value_b   = (const float*)d_in[7];
  const float* proj_w    = (const float*)d_in[8];
  const float* proj_b    = (const float*)d_in[9];
  const float* mlp_w1    = (const float*)d_in[10];
  const float* mlp_b1    = (const float*)d_in[11];
  const float* mlp_w2    = (const float*)d_in[12];
  const float* mlp_b2    = (const float*)d_in[13];
  float* outp = (float*)d_out;
  char*  ws   = (char*)d_ws;

  // Workspace layout (bytes).  hidden aliases value+attnsum (both dead then).
  constexpr size_t OFF_W_OA   = 0;                                  // 96*2304 bf16
  constexpr size_t OFF_W_VAL  = OFF_W_OA  + 96u  * 2304u * 2u;
  constexpr size_t OFF_W_PROJ = OFF_W_VAL + 256u * 256u  * 2u;
  constexpr size_t OFF_W_M1   = OFF_W_PROJ+ 256u * 256u  * 2u;
  constexpr size_t OFF_W_M2   = OFF_W_M1  + 512u * 256u  * 2u;
  constexpr size_t OFF_BIAS   = OFF_W_M2  + 256u * 512u  * 2u;      // 96 f32
  constexpr size_t OFF_OA     = (OFF_BIAS + 384u + 255u) & ~size_t(255);
  constexpr size_t OFF_VALUE  = OFF_OA    + 8u * 96u  * 4096u * 4u;
  constexpr size_t OFF_ATTNS  = OFF_VALUE + 8u * 256u * 4096u * 4u;
  constexpr size_t OFF_PROJ   = OFF_ATTNS + 8u * 256u * 4096u * 4u;

  unsigned short* wOA   = (unsigned short*)(ws + OFF_W_OA);
  unsigned short* wVAL  = (unsigned short*)(ws + OFF_W_VAL);
  unsigned short* wPROJ = (unsigned short*)(ws + OFF_W_PROJ);
  unsigned short* wM1   = (unsigned short*)(ws + OFF_W_M1);
  unsigned short* wM2   = (unsigned short*)(ws + OFF_W_M2);
  float* biasOA  = (float*)(ws + OFF_BIAS);
  float* offattn = (float*)(ws + OFF_OA);
  float* value   = (float*)(ws + OFF_VALUE);
  float* attnsum = (float*)(ws + OFF_ATTNS);
  float* hidden  = (float*)(ws + OFF_VALUE);   // aliases value+attnsum (67 MB)
  float* projout = (float*)(ws + OFF_PROJ);

  // --- weight prep (f32 -> bf16, conv weights reordered r-major) ---
  prep_offattn_w_kernel<<<(96 * 2304 + 255) / 256, 256, 0, stream>>>(offset_w, attn_w, wOA);
  prep_offattn_b_kernel<<<1, 96, 0, stream>>>(offset_b, attn_b, biasOA);
  cvt_bf16_kernel<<<(65536  + 255) / 256, 256, 0, stream>>>(value_w, wVAL, 65536);
  cvt_bf16_kernel<<<(65536  + 255) / 256, 256, 0, stream>>>(proj_w,  wPROJ, 65536);
  cvt_bf16_kernel<<<(131072 + 255) / 256, 256, 0, stream>>>(mlp_w1,  wM1, 131072);
  cvt_bf16_kernel<<<(131072 + 255) / 256, 256, 0, stream>>>(mlp_w2,  wM2, 131072);

  dim3 blk(256);
  // 1) fused offsets+attn 3x3 conv (implicit GEMM, Cout=96, K=9*256 r-major)
  gemm_wmma<1, 0, 0><<<dim3(32, 1, 8), blk, 0, stream>>>(wOA, f_recalib, biasOA, nullptr, offattn, 96, 2304, 256);
  // 2) value 1x1 conv
  gemm_wmma<0, 0, 0><<<dim3(32, 2, 8), blk, 0, stream>>>(wVAL, f_recalib, value_b, nullptr, value, 256, 256, 256);
  // 3) softmax + grid-sample + attention-weighted sum
  sample_attn_kernel<<<(8 * 8 * 4096) / 256, 256, 0, stream>>>(offattn, value, attnsum);
  // 4) proj 1x1 + residual f_orig
  gemm_wmma<0, 0, 1><<<dim3(32, 2, 8), blk, 0, stream>>>(wPROJ, attnsum, proj_b, f_orig, projout, 256, 256, 256);
  // 5) mlp1 (256->512) + ReLU
  gemm_wmma<0, 1, 0><<<dim3(32, 4, 8), blk, 0, stream>>>(wM1, projout, mlp_b1, nullptr, hidden, 512, 256, 256);
  // 6) mlp2 (512->256) + residual projout -> d_out
  gemm_wmma<0, 0, 1><<<dim3(32, 2, 8), blk, 0, stream>>>(wM2, hidden, mlp_b2, projout, outp, 256, 512, 512);
}
